// SpaceTimeTransformer_50903952392594
// MI455X (gfx1250) — matrix operand order repaired and checked by reading, not compiled, not run on hardware
//
#include <hip/hip_runtime.h>
#include <cstdint>
#include <cstddef>

// ---------------- model constants ----------------
constexpr int Bc   = 4;
constexpr int Fc   = 8;
constexpr int Pc   = 196;
constexpr int Dc   = 768;
constexpr int Hc   = 12;          // heads
constexpr int HDc  = 64;          // head dim
constexpr int Lc   = 4;
constexpr int MLPc = 3072;
constexpr int Nc   = 1 + Fc * Pc; // 1569 tokens
constexpr int Mc   = Bc * Nc;     // 6276 rows
constexpr float SCALEc = 0.125f;  // 64^-0.5

typedef __bf16 bhalf;
typedef __attribute__((ext_vector_type(16))) __bf16 v16bf;
typedef __attribute__((ext_vector_type(8)))  float  v8f;
typedef __attribute__((ext_vector_type(4)))  int    v4i;

// ---- CDNA5 async global->LDS copy path (probe via __has_builtin) ----
#if defined(__has_builtin)
#  if __has_builtin(__builtin_amdgcn_global_load_async_to_lds_b128)
#    define USE_ASYNC_LDS 1
#  endif
#endif

#if defined(USE_ASYNC_LDS)
__device__ __forceinline__ void async_copy16(const void* gsrc, void* ldst) {
    __builtin_amdgcn_global_load_async_to_lds_b128((v4i*)gsrc, (v4i*)ldst, 0, 0);
}
__device__ __forceinline__ void wait_async0() {
#  if __has_builtin(__builtin_amdgcn_s_wait_asynccnt)
    __builtin_amdgcn_s_wait_asynccnt(0);
#  else
    asm volatile("s_wait_asynccnt 0" ::: "memory");
#  endif
}
#endif

// ---------------- fp32 -> bf16 convert ----------------
__global__ void convert_bf16_kernel(const float* __restrict__ s, bhalf* __restrict__ d, long long n) {
    long long i = (long long)blockIdx.x * blockDim.x + threadIdx.x;
    long long stride = (long long)gridDim.x * blockDim.x;
    for (; i < n; i += stride) d[i] = (bhalf)s[i];
}

// ---------------- LayerNorm (row of 768) -> bf16 ----------------
__global__ __launch_bounds__(256) void ln_bf16_kernel(const float* __restrict__ x,
                                                      const float* __restrict__ g,
                                                      const float* __restrict__ beta,
                                                      bhalf* __restrict__ out) {
    int r = blockIdx.x, tid = threadIdx.x;
    __shared__ float red[256];
    const float* xr = x + (size_t)r * Dc;
    float vals[3], s = 0.f;
#pragma unroll
    for (int i = 0; i < 3; ++i) { vals[i] = xr[tid + i * 256]; s += vals[i]; }
    red[tid] = s; __syncthreads();
    for (int st = 128; st > 0; st >>= 1) { if (tid < st) red[tid] += red[tid + st]; __syncthreads(); }
    float mean = red[0] * (1.f / Dc); __syncthreads();
    float s2 = 0.f;
#pragma unroll
    for (int i = 0; i < 3; ++i) { float dv = vals[i] - mean; s2 += dv * dv; }
    red[tid] = s2; __syncthreads();
    for (int st = 128; st > 0; st >>= 1) { if (tid < st) red[tid] += red[tid + st]; __syncthreads(); }
    float rstd = rsqrtf(red[0] * (1.f / Dc) + 1e-5f);
#pragma unroll
    for (int i = 0; i < 3; ++i) {
        int d = tid + i * 256;
        out[(size_t)r * Dc + d] = (bhalf)((vals[i] - mean) * rstd * g[d] + beta[d]);
    }
}

// ---------------- WMMA bf16 GEMM: C = A(MxK) * Bw(NdxK)^T ----------------
// Block tile 128x128, 8 waves (4x2), wave tile 32x64 (2x4 WMMA accs).
// Double-buffered LDS; staging via CDNA5 async global->LDS copies when
// available (ASYNCcnt), else global->reg->ds_store.
// EPI 0: store f32             (qkv, no bias)
// EPI 1: f32 + bias + residual (proj / fc2)
// EPI 2: bias + exact GELU -> bf16 (fc1)
template<int EPI>
__global__ __launch_bounds__(256) void gemm_bf16_kernel(const bhalf* __restrict__ A,
                                                        const bhalf* __restrict__ Bw,
                                                        const float* __restrict__ bias,
                                                        const float* __restrict__ res,
                                                        float* __restrict__ outF,
                                                        bhalf* __restrict__ outB,
                                                        int M, int Nd, int K) {
    constexpr int LDSS = 40; // halves; 16B-aligned rows, staggers banks
    __shared__ __align__(16) bhalf sA[2][128 * LDSS];
    __shared__ __align__(16) bhalf sB[2][128 * LDSS];

    const int tid  = threadIdx.x;
    const int lane = tid & 31, wave = tid >> 5;
    const int wm = wave >> 1, wn = wave & 1;          // 4 x 2 wave grid
    const int hi = lane >> 4, l15 = lane & 15;
    const int m0 = blockIdx.x * 128, n0 = blockIdx.y * 128;

    // staging chunk assignment: 512 16B-chunks per matrix, 2 per thread each
    const int row0 = tid >> 2, seg = tid & 3;  // rows 0..63
    const int row1 = row0 + 64;                // rows 64..127
    const bool gA0 = (m0 + row0) < M;
    const bool gA1 = (m0 + row1) < M;
    const bhalf* a0p = A  + (size_t)(m0 + row0) * K + seg * 8;
    const bhalf* a1p = A  + (size_t)(m0 + row1) * K + seg * 8;
    const bhalf* b0p = Bw + (size_t)(n0 + row0) * K + seg * 8;
    const bhalf* b1p = Bw + (size_t)(n0 + row1) * K + seg * 8;
    const int sOff0 = row0 * LDSS + seg * 8;
    const int sOff1 = row1 * LDSS + seg * 8;

    v8f acc[2][4];
#pragma unroll
    for (int ti = 0; ti < 2; ++ti)
#pragma unroll
    for (int tj = 0; tj < 4; ++tj) acc[ti][tj] = (v8f){0, 0, 0, 0, 0, 0, 0, 0};

#if defined(USE_ASYNC_LDS)
    // prologue: async-stage K-tile 0 into buffer 0 (no VGPR round-trip)
    if (gA0) async_copy16(a0p, &sA[0][sOff0]);
    if (gA1) async_copy16(a1p, &sA[0][sOff1]);
    async_copy16(b0p, &sB[0][sOff0]);
    async_copy16(b1p, &sB[0][sOff1]);
    wait_async0();
#else
    const uint4 z4 = {0u, 0u, 0u, 0u};
    uint4 rA0, rA1, rB0, rB1;
    rA0 = gA0 ? *reinterpret_cast<const uint4*>(a0p) : z4;
    rA1 = gA1 ? *reinterpret_cast<const uint4*>(a1p) : z4;
    rB0 = *reinterpret_cast<const uint4*>(b0p);
    rB1 = *reinterpret_cast<const uint4*>(b1p);
    *reinterpret_cast<uint4*>(&sA[0][sOff0]) = rA0;
    *reinterpret_cast<uint4*>(&sA[0][sOff1]) = rA1;
    *reinterpret_cast<uint4*>(&sB[0][sOff0]) = rB0;
    *reinterpret_cast<uint4*>(&sB[0][sOff1]) = rB1;
#endif
    __syncthreads();

    for (int kk = 0; kk < K; kk += 32) {
        const int cur = (kk >> 5) & 1;
        const bool hasNext = (kk + 32) < K;
        const int nxt = cur ^ 1;

        // issue next tile's copies early (latency hidden under 8 WMMAs)
        if (hasNext) {
            const int kn = kk + 32;
#if defined(USE_ASYNC_LDS)
            if (gA0) async_copy16(a0p + kn, &sA[nxt][sOff0]);
            if (gA1) async_copy16(a1p + kn, &sA[nxt][sOff1]);
            async_copy16(b0p + kn, &sB[nxt][sOff0]);
            async_copy16(b1p + kn, &sB[nxt][sOff1]);
#else
            rA0 = gA0 ? *reinterpret_cast<const uint4*>(a0p + kn) : z4;
            rA1 = gA1 ? *reinterpret_cast<const uint4*>(a1p + kn) : z4;
            rB0 = *reinterpret_cast<const uint4*>(b0p + kn);
            rB1 = *reinterpret_cast<const uint4*>(b1p + kn);
#endif
        }

        // fragment loads from current buffer + 8 WMMAs
        const bhalf* bufA = sA[cur];
        const bhalf* bufB = sB[cur];
        union Frag { uint4 u[2]; v16bf v; } fa[2], fb[4];
#pragma unroll
        for (int ti = 0; ti < 2; ++ti) {
            // A frag (ISA 7.12.2, 16-bit A 16x32): k = hi*8..+7 then 16+hi*8..+7
            int r = (wm * 32 + ti * 16 + l15) * LDSS;
            fa[ti].u[0] = *reinterpret_cast<const uint4*>(&bufA[r + hi * 8]);
            fa[ti].u[1] = *reinterpret_cast<const uint4*>(&bufA[r + 16 + hi * 8]);
        }
#pragma unroll
        for (int tj = 0; tj < 4; ++tj) {
            // B frag: column n = lane&15, k = hi*8..+7 then 16+hi*8..+7
            int c = (wn * 64 + tj * 16 + l15) * LDSS;
            fb[tj].u[0] = *reinterpret_cast<const uint4*>(&bufB[c + hi * 8]);
            fb[tj].u[1] = *reinterpret_cast<const uint4*>(&bufB[c + 16 + hi * 8]);
        }
#pragma unroll
        for (int ti = 0; ti < 2; ++ti)
#pragma unroll
        for (int tj = 0; tj < 4; ++tj)
            acc[ti][tj] = __builtin_amdgcn_wmma_f32_16x16x32_bf16(
                false, fa[ti].v, false, fb[tj].v, (short)0, acc[ti][tj], false, false);

        // retire next-tile staging; single barrier per iteration
        if (hasNext) {
#if defined(USE_ASYNC_LDS)
            wait_async0();
#else
            *reinterpret_cast<uint4*>(&sA[nxt][sOff0]) = rA0;
            *reinterpret_cast<uint4*>(&sA[nxt][sOff1]) = rA1;
            *reinterpret_cast<uint4*>(&sB[nxt][sOff0]) = rB0;
            *reinterpret_cast<uint4*>(&sB[nxt][sOff1]) = rB1;
#endif
        }
        __syncthreads();
    }

    // epilogue
#pragma unroll
    for (int ti = 0; ti < 2; ++ti)
#pragma unroll
    for (int tj = 0; tj < 4; ++tj) {
#pragma unroll
        for (int i = 0; i < 8; ++i) {
            int row = m0 + wm * 32 + ti * 16 + hi * 8 + i;  // C layout: M = vgpr + hi*8
            int col = n0 + wn * 64 + tj * 16 + l15;         // N = lane&15
            if (row < M) {
                float v = acc[ti][tj][i];
                if (EPI == 0) {
                    outF[(size_t)row * Nd + col] = v;
                } else if (EPI == 1) {
                    v += bias[col] + res[(size_t)row * Nd + col];
                    outF[(size_t)row * Nd + col] = v;
                } else {
                    v += bias[col];
                    float ge = 0.5f * v * (1.0f + erff(v * 0.70710678f));
                    outB[(size_t)row * Nd + col] = (bhalf)ge;
                }
            }
        }
    }
}

// ---------------- cls-token attention (full softmax over all N tokens) ----------------
__global__ __launch_bounds__(256) void attn_cls_kernel(const float* __restrict__ qkv, bhalf* __restrict__ obf) {
    int bh = blockIdx.x, b = bh / Hc, h = bh % Hc, tid = threadIdx.x;
    __shared__ float qs[HDc];
    __shared__ float sim[Nc];
    __shared__ float red[256];
    const float* base = qkv + (size_t)b * Nc * (3 * Dc);
    if (tid < HDc) qs[tid] = base[h * HDc + tid] * SCALEc;
    __syncthreads();
    float lmax = -3.0e38f;
    for (int n = tid; n < Nc; n += 256) {
        const float* kr = base + (size_t)n * (3 * Dc) + Dc + h * HDc;
        float d = 0.f;
#pragma unroll
        for (int i = 0; i < HDc; ++i) d += qs[i] * kr[i];
        sim[n] = d; lmax = fmaxf(lmax, d);
    }
    red[tid] = lmax; __syncthreads();
    for (int st = 128; st > 0; st >>= 1) { if (tid < st) red[tid] = fmaxf(red[tid], red[tid + st]); __syncthreads(); }
    float m = red[0]; __syncthreads();
    float ls = 0.f;
    for (int n = tid; n < Nc; n += 256) { float e = expf(sim[n] - m); sim[n] = e; ls += e; }
    red[tid] = ls; __syncthreads();
    for (int st = 128; st > 0; st >>= 1) { if (tid < st) red[tid] += red[tid + st]; __syncthreads(); }
    float ssum = red[0]; __syncthreads();
    if (tid < HDc) {
        float o = 0.f;
        for (int n = 0; n < Nc; ++n)
            o += sim[n] * base[(size_t)n * (3 * Dc) + 2 * Dc + h * HDc + tid];
        obf[(size_t)(b * Nc) * Dc + h * HDc + tid] = (bhalf)(o / ssum);
    }
}

// ---------------- time attention: groups (b,h,p), F queries over F+1 keys ----------------
__global__ __launch_bounds__(256) void attn_time_kernel(const float* __restrict__ qkv, bhalf* __restrict__ obf) {
    int gid = blockIdx.x * 256 + threadIdx.x;
    if (gid >= Bc * Hc * Pc * Fc) return;
    int f = gid % Fc;
    int p = (gid / Fc) % Pc;
    int bh = gid / (Fc * Pc);
    int b = bh / Hc, h = bh % Hc;
    const float* base = qkv + (size_t)b * Nc * (3 * Dc);
    int nq = 1 + f * Pc + p;
    float q[HDc];
    const float* qr = base + (size_t)nq * (3 * Dc) + h * HDc;
#pragma unroll
    for (int i = 0; i < HDc; ++i) q[i] = qr[i] * SCALEc;
    int nk[Fc + 1];
    nk[0] = 0;
#pragma unroll
    for (int j = 1; j <= Fc; ++j) nk[j] = 1 + (j - 1) * Pc + p;
    float a[Fc + 1], m = -3.0e38f;
#pragma unroll
    for (int j = 0; j <= Fc; ++j) {
        const float* kr = base + (size_t)nk[j] * (3 * Dc) + Dc + h * HDc;
        float d = 0.f;
#pragma unroll
        for (int i = 0; i < HDc; ++i) d += q[i] * kr[i];
        a[j] = d; m = fmaxf(m, d);
    }
    float s = 0.f;
#pragma unroll
    for (int j = 0; j <= Fc; ++j) { a[j] = expf(a[j] - m); s += a[j]; }
    float inv = 1.f / s;
    bhalf* orow = obf + (size_t)(b * Nc + nq) * Dc + h * HDc;
#pragma unroll
    for (int i = 0; i < HDc; ++i) {
        float o = 0.f;
#pragma unroll
        for (int j = 0; j <= Fc; ++j)
            o += a[j] * base[(size_t)nk[j] * (3 * Dc) + 2 * Dc + h * HDc + i];
        orow[i] = (bhalf)(o * inv);
    }
}

// ---------------- space attention: groups (b,h,f), P queries over P+1 keys ----------------
// one wave per query (online softmax), keys staged in LDS
__global__ __launch_bounds__(256) void attn_space_kernel(const float* __restrict__ qkv, bhalf* __restrict__ obf) {
    int grp = blockIdx.x;                 // Bc*Hc*Fc = 384
    int f = grp % Fc, bh = grp / Fc;
    int b = bh / Hc, h = bh % Hc;
    int tid = threadIdx.x, lane = tid & 31, wave = tid >> 5;
    __shared__ float ks[(Pc + 1) * HDc];  // 197*64 f32 = 50.4 KB
    const float* base = qkv + (size_t)b * Nc * (3 * Dc);
    for (int idx = tid; idx < (Pc + 1) * HDc; idx += 256) {
        int j = idx >> 6, d = idx & 63;
        int n = (j == 0) ? 0 : (1 + f * Pc + (j - 1));
        ks[idx] = base[(size_t)n * (3 * Dc) + Dc + h * HDc + d];
    }
    __syncthreads();
    int d0 = lane * 2;
    for (int p = wave; p < Pc; p += 8) {
        int nq = 1 + f * Pc + p;
        const float* qr = base + (size_t)nq * (3 * Dc) + h * HDc;
        float q0 = qr[d0] * SCALEc, q1 = qr[d0 + 1] * SCALEc;
        float m = -3.0e38f, s = 0.f, a0 = 0.f, a1 = 0.f;
        for (int j = 0; j <= Pc; ++j) {
            float part = q0 * ks[j * HDc + d0] + q1 * ks[j * HDc + d0 + 1];
#pragma unroll
            for (int off = 16; off > 0; off >>= 1) part += __shfl_xor(part, off, 32);
            float mn = fmaxf(m, part);
            float corr = expf(m - mn);
            float e = expf(part - mn);
            s = s * corr + e;
            a0 *= corr; a1 *= corr;
            int n = (j == 0) ? 0 : (1 + f * Pc + (j - 1));
            const float* vr = base + (size_t)n * (3 * Dc) + 2 * Dc + h * HDc;
            a0 += e * vr[d0]; a1 += e * vr[d0 + 1];
            m = mn;
        }
        float inv = 1.f / s;
        bhalf* orow = obf + (size_t)(b * Nc + nq) * Dc + h * HDc;
        orow[d0]     = (bhalf)(a0 * inv);
        orow[d0 + 1] = (bhalf)(a1 * inv);
    }
}

// ---------------- host orchestration ----------------
static inline void* wsAlloc(char* base, size_t& off, size_t bytes) {
    off = (off + 255) & ~(size_t)255;
    void* p = base + off;
    off += bytes;
    return p;
}

extern "C" void kernel_launch(void* const* d_in, const int* in_sizes, int n_in,
                              void* d_out, int out_size, void* d_ws, size_t ws_size,
                              hipStream_t stream) {
    (void)in_sizes; (void)n_in; (void)out_size; (void)ws_size;
    const float* x0       = (const float*)d_in[0];
    const float* qkv_w_t  = (const float*)d_in[1];
    const float* proj_w_t = (const float*)d_in[2];
    const float* proj_b_t = (const float*)d_in[3];
    const float* qkv_w_s  = (const float*)d_in[4];
    const float* proj_w_s = (const float*)d_in[5];
    const float* proj_b_s = (const float*)d_in[6];
    const float* fc1_w    = (const float*)d_in[7];
    const float* fc1_b    = (const float*)d_in[8];
    const float* fc2_w    = (const float*)d_in[9];
    const float* fc2_b    = (const float*)d_in[10];
    const float* ln1_g    = (const float*)d_in[11];
    const float* ln1_b    = (const float*)d_in[12];
    const float* ln2_g    = (const float*)d_in[13];
    const float* ln2_b    = (const float*)d_in[14];
    const float* ln3_g    = (const float*)d_in[15];
    const float* ln3_b    = (const float*)d_in[16];

    char* ws = (char*)d_ws; size_t off = 0;
    bhalf* wQKVt = (bhalf*)wsAlloc(ws, off, (size_t)Lc * 3 * Dc * Dc * 2);
    bhalf* wPRJt = (bhalf*)wsAlloc(ws, off, (size_t)Lc * Dc * Dc * 2);
    bhalf* wQKVs = (bhalf*)wsAlloc(ws, off, (size_t)Lc * 3 * Dc * Dc * 2);
    bhalf* wPRJs = (bhalf*)wsAlloc(ws, off, (size_t)Lc * Dc * Dc * 2);
    bhalf* wFC1  = (bhalf*)wsAlloc(ws, off, (size_t)Lc * MLPc * Dc * 2);
    bhalf* wFC2  = (bhalf*)wsAlloc(ws, off, (size_t)Lc * Dc * MLPc * 2);
    bhalf* Abf   = (bhalf*)wsAlloc(ws, off, (size_t)Mc * Dc * 2);
    float* QKV   = (float*)wsAlloc(ws, off, (size_t)Mc * 3 * Dc * 4);
    bhalf* Obf   = (bhalf*)wsAlloc(ws, off, (size_t)Mc * Dc * 2);
    bhalf* H1    = (bhalf*)wsAlloc(ws, off, (size_t)Mc * MLPc * 2);
    float* Tres  = (float*)wsAlloc(ws, off, (size_t)Mc * Dc * 4);
    float* Sres  = (float*)wsAlloc(ws, off, (size_t)Mc * Dc * 4);
    float* Xa    = (float*)wsAlloc(ws, off, (size_t)Mc * Dc * 4);
    float* Xb    = (float*)wsAlloc(ws, off, (size_t)Mc * Dc * 4);

    auto cvt = [&](const float* s, bhalf* d, size_t n) {
        convert_bf16_kernel<<<2048, 256, 0, stream>>>(s, d, (long long)n);
    };
    cvt(qkv_w_t,  wQKVt, (size_t)Lc * 3 * Dc * Dc);
    cvt(proj_w_t, wPRJt, (size_t)Lc * Dc * Dc);
    cvt(qkv_w_s,  wQKVs, (size_t)Lc * 3 * Dc * Dc);
    cvt(proj_w_s, wPRJs, (size_t)Lc * Dc * Dc);
    cvt(fc1_w,    wFC1,  (size_t)Lc * MLPc * Dc);
    cvt(fc2_w,    wFC2,  (size_t)Lc * Dc * MLPc);

    const int MT = (Mc + 127) / 128;         // 50
    const dim3 blk(256);

    for (int l = 0; l < Lc; ++l) {
        const float* xin = (l == 0) ? x0 : ((l & 1) ? Xa : Xb);
        float* xout = (l == 3) ? (float*)d_out : ((l & 1) ? Xb : Xa);
        const bhalf* wqt = wQKVt + (size_t)l * 3 * Dc * Dc;
        const bhalf* wpt = wPRJt + (size_t)l * Dc * Dc;
        const bhalf* wqs = wQKVs + (size_t)l * 3 * Dc * Dc;
        const bhalf* wps = wPRJs + (size_t)l * Dc * Dc;
        const bhalf* wf1 = wFC1  + (size_t)l * MLPc * Dc;
        const bhalf* wf2 = wFC2  + (size_t)l * Dc * MLPc;

        // -------- time attention branch --------
        ln_bf16_kernel<<<Mc, blk, 0, stream>>>(xin, ln3_g + l * Dc, ln3_b + l * Dc, Abf);
        gemm_bf16_kernel<0><<<dim3(MT, 18), blk, 0, stream>>>(Abf, wqt, nullptr, nullptr, QKV, nullptr, Mc, 3 * Dc, Dc);
        attn_cls_kernel<<<Bc * Hc, blk, 0, stream>>>(QKV, Obf);
        attn_time_kernel<<<(Bc * Hc * Pc * Fc + 255) / 256, blk, 0, stream>>>(QKV, Obf);
        gemm_bf16_kernel<1><<<dim3(MT, 6), blk, 0, stream>>>(Obf, wpt, proj_b_t + l * Dc, xin, Tres, nullptr, Mc, Dc, Dc);

        // -------- space attention branch (residual from xin, per reference) --------
        ln_bf16_kernel<<<Mc, blk, 0, stream>>>(Tres, ln1_g + l * Dc, ln1_b + l * Dc, Abf);
        gemm_bf16_kernel<0><<<dim3(MT, 18), blk, 0, stream>>>(Abf, wqs, nullptr, nullptr, QKV, nullptr, Mc, 3 * Dc, Dc);
        attn_cls_kernel<<<Bc * Hc, blk, 0, stream>>>(QKV, Obf);
        attn_space_kernel<<<Bc * Hc * Fc, blk, 0, stream>>>(QKV, Obf);
        gemm_bf16_kernel<1><<<dim3(MT, 6), blk, 0, stream>>>(Obf, wps, proj_b_s + l * Dc, xin, Sres, nullptr, Mc, Dc, Dc);

        // -------- MLP --------
        ln_bf16_kernel<<<Mc, blk, 0, stream>>>(Sres, ln2_g + l * Dc, ln2_b + l * Dc, Abf);
        gemm_bf16_kernel<2><<<dim3(MT, 24), blk, 0, stream>>>(Abf, wf1, fc1_b + l * MLPc, nullptr, nullptr, H1, Mc, MLPc, Dc);
        gemm_bf16_kernel<1><<<dim3(MT, 6), blk, 0, stream>>>(H1, wf2, fc2_b + l * Dc, Sres, xout, nullptr, Mc, Dc, MLPc);
    }
}
